// STA_SelfAttention_38070590112266
// MI455X (gfx1250) — matrix operand order repaired and checked by reading, not compiled, not run on hardware
//
#include <hip/hip_runtime.h>

// ---------------------------------------------------------------------------
// STA self-attention forward for MI455X (gfx1250, wave32, WMMA).
// All contractions via v_wmma_f32_16x16x32_bf16 (bf16 in, f32 accumulate).
// ---------------------------------------------------------------------------

typedef __bf16 bf16;
typedef __attribute__((ext_vector_type(16))) bf16  v16bf;
typedef __attribute__((ext_vector_type(8)))  bf16  v8bf;
typedef __attribute__((ext_vector_type(8)))  float v8f;

#define S_  4
#define T_  4096
#define D_  1024
#define TB_ 1024
#define WB_ 256
#define NB_ 16   // S*S blocks

// f32 -> bf16 with round-to-nearest-even (bit trick: no reliance on __bf16 arith)
__device__ __forceinline__ bf16 f2bf(float f) {
  union { float f; unsigned u; } v; v.f = f;
  unsigned r = v.u + 0x7FFFu + ((v.u >> 16) & 1u);
  unsigned short h = (unsigned short)(r >> 16);
  return __builtin_bit_cast(bf16, h);
}

// -------- WMMA fragment loads (ISA 7.12.2 layouts, wave32) ------------------
// A fragment: 16x32 bf16 tile, row-major source, leading dim lda.
// lane<16: M=lane, K = {0..7, 16..23}; lane>=16: M=lane-16, K = {8..15, 24..31}
__device__ __forceinline__ v16bf frag_a(const bf16* __restrict__ A, int lda) {
  const int lane = threadIdx.x & 31;
  const int m  = lane & 15;
  const int kb = (lane < 16) ? 0 : 8;
  const bf16* p = A + (size_t)m * lda + kb;
  v8bf lo = *(const v8bf*)(p);
  v8bf hi = *(const v8bf*)(p + 16);
  return __builtin_shufflevector(lo, hi, 0,1,2,3,4,5,6,7,8,9,10,11,12,13,14,15);
}

// B fragment: 32x16 bf16 (KxN) where memory holds operand as [N,K] row-major
// (i.e. C = A * B^T). lane<16: N=lane, K = {0..7, 16..23}; lane>=16: N=lane-16, K = {8..15, 24..31}.
__device__ __forceinline__ v16bf frag_b(const bf16* __restrict__ B, int ldb) {
  const int lane = threadIdx.x & 31;
  const int n  = lane & 15;
  const int kb = (lane < 16) ? 0 : 8;
  const bf16* p = B + (size_t)n * ldb + kb;
  v8bf lo = *(const v8bf*)(p);
  v8bf hi = *(const v8bf*)(p + 16);
  return __builtin_shufflevector(lo, hi, 0,1,2,3,4,5,6,7,8,9,10,11,12,13,14,15);
}

// -------- batched NT GEMM: C[z] = A[z] * B[z]^T (+epilogue) -----------------
// Batch index z = blockIdx.z decomposed as (i = z>>2, j = z&3); every operand
// pointer is offset by i*s?i + j*s?j, which covers: no batch (z=1), per-row-
// block batch (z=4, use s?j), and the full 16-block (i,j) grid.
// MODE: 0 = f32 out (+optional f32 addend matrix)
//       1 = bf16 out
//       2 = bf16 out, stored transposed (C[n*ldc+m])
//       3 = f32 out, val = 1 - val            (diversity)
//       5 = bf16 out, val += rv0[m]*cv0[n*cs] + rv1[m]*cv1[n*cs]
template<int MODE>
__global__ void __launch_bounds__(128)
gemm_bf16_nt(const bf16* __restrict__ A, int lda, long long sAi, long long sAj,
             const bf16* __restrict__ B, int ldb, long long sBi, long long sBj,
             float* __restrict__ Cf, bf16* __restrict__ Cb,
             int ldc, long long sCi, long long sCj,
             int M, int N, int K,
             const float* __restrict__ addend, int ld_add,
             const float* __restrict__ rowv0, const float* __restrict__ rowv1,
             long long sRi, long long sRj,
             const float* __restrict__ colv0, const float* __restrict__ colv1,
             int colstride)
{
  const int z  = blockIdx.z;
  const int bi = z >> 2, bj = z & 3;
  A += bi * sAi + bj * sAj;
  B += bi * sBi + bj * sBj;
  const long long coff = bi * sCi + bj * sCj;
  if (Cf) Cf += coff;
  if (Cb) Cb += coff;
  const float* rv0 = rowv0 ? rowv0 + bi * sRi + bj * sRj : nullptr;
  const float* rv1 = rowv1 ? rowv1 + bi * sRi + bj * sRj : nullptr;

  // 4 waves: 2x2 grid of 32x64 wave tiles -> 64(M) x 128(N) block tile.
  const int wave = threadIdx.x >> 5;
  const int wm = wave & 1, wn = wave >> 1;
  const int m0 = blockIdx.y * 64  + wm * 32;
  const int n0 = blockIdx.x * 128 + wn * 64;

  v8f acc[2][4];
#pragma unroll
  for (int tm = 0; tm < 2; ++tm)
#pragma unroll
    for (int tn = 0; tn < 4; ++tn) acc[tm][tn] = (v8f)(0.0f);

  for (int k = 0; k < K; k += 32) {
    v16bf af[2];
    af[0] = frag_a(A + (size_t)(m0)      * lda + k, lda);
    af[1] = frag_a(A + (size_t)(m0 + 16) * lda + k, lda);
    v16bf bfr[4];
#pragma unroll
    for (int tn = 0; tn < 4; ++tn)
      bfr[tn] = frag_b(B + (size_t)(n0 + 16 * tn) * ldb + k, ldb);
#pragma unroll
    for (int tm = 0; tm < 2; ++tm)
#pragma unroll
      for (int tn = 0; tn < 4; ++tn)
        acc[tm][tn] = __builtin_amdgcn_wmma_f32_16x16x32_bf16(
            false, af[tm], false, bfr[tn], (short)0, acc[tm][tn], false, false);
  }

  // C fragment layout: lane<16 -> M = v, lane>=16 -> M = v+8; N = lane&15.
  const int lane = threadIdx.x & 31;
  const int cn = lane & 15;
  const int mb = (lane < 16) ? 0 : 8;
#pragma unroll
  for (int tm = 0; tm < 2; ++tm) {
#pragma unroll
    for (int tn = 0; tn < 4; ++tn) {
#pragma unroll
      for (int v = 0; v < 8; ++v) {
        const int row = m0 + tm * 16 + mb + v;
        const int col = n0 + tn * 16 + cn;
        float c = acc[tm][tn][v];
        if constexpr (MODE == 0) {
          if (addend) c += addend[(size_t)row * ld_add + col];
          Cf[(size_t)row * ldc + col] = c;
        } else if constexpr (MODE == 1) {
          Cb[(size_t)row * ldc + col] = f2bf(c);
        } else if constexpr (MODE == 2) {
          Cb[(size_t)col * ldc + row] = f2bf(c);
        } else if constexpr (MODE == 3) {
          Cf[(size_t)row * ldc + col] = 1.0f - c;
        } else if constexpr (MODE == 5) {
          c += rv0[row] * colv0[(size_t)col * colstride]
             + rv1[row] * colv1[(size_t)col * colstride];
          Cb[(size_t)row * ldc + col] = f2bf(c);
        }
      }
    }
  }
}

// -------- elementwise / reduction kernels -----------------------------------
__global__ void __launch_bounds__(256)
rownorm_kernel(const float* __restrict__ x, bf16* __restrict__ xbf,
               bf16* __restrict__ xu)
{
  __shared__ float red[256];
  const int t = blockIdx.x, tid = threadIdx.x;
  const float* row = x + (size_t)t * D_;
  float s = 0.f;
  for (int c = tid; c < D_; c += 256) { float v = row[c]; s += v * v; }
  red[tid] = s; __syncthreads();
  for (int o = 128; o > 0; o >>= 1) { if (tid < o) red[tid] += red[tid + o]; __syncthreads(); }
  const float scale = 1.0f / fmaxf(sqrtf(red[0]), 1e-12f);
  for (int c = tid; c < D_; c += 256) {
    float v = row[c];
    xbf[(size_t)t * D_ + c] = f2bf(v);
    xu [(size_t)t * D_ + c] = f2bf(v * scale);
  }
}

__global__ void __launch_bounds__(256)
convert_kernel(const float* __restrict__ src, bf16* __restrict__ dst, int n)
{
  int i = blockIdx.x * 256 + threadIdx.x;
  if (i < n) dst[i] = f2bf(src[i]);
}

__global__ void __launch_bounds__(256)
w2a_kernel(const float* __restrict__ w2, bf16* __restrict__ dst)  // 256x256 of [256,258]
{
  int i = blockIdx.x * 256 + threadIdx.x;   // 65536
  int r = i >> 8, c = i & 255;
  dst[i] = f2bf(w2[r * 258 + c]);
}

__global__ void __launch_bounds__(256)
ap_kernel(float* __restrict__ ap)   // AP[q,k], 1024x1024
{
  int i = blockIdx.x * 256 + threadIdx.x;   // 1M
  int q = i >> 10, k = i & 1023;
  int h = k >> 1;
  float inv = expf(-(2.0f * (float)h / (float)D_) * logf(10000.0f));
  float ang = (float)q * inv;
  ap[i] = (k & 1) ? cosf(ang) : sinf(ang);
}

// per (block b, query q): softmax stats + entropy + diversity dependency
__global__ void __launch_bounds__(256)
softstat_kernel(const float* __restrict__ sim, const float* __restrict__ Dd,
                float* __restrict__ ent_raw, float* __restrict__ dep,
                float* __restrict__ rowm, float* __restrict__ rowrz)
{
  __shared__ float  red[256];
  __shared__ float4 red4[256];
  const int q = blockIdx.x, b = blockIdx.y, tid = threadIdx.x;
  const int i = b >> 2;                                  // row-block index
  const float* srow = sim + ((size_t)b * TB_ + q) * TB_;
  const float* drow = Dd  + ((size_t)i * TB_ + q) * TB_;
  float m = -3.4e38f;
  for (int k = tid; k < TB_; k += 256) m = fmaxf(m, srow[k]);
  red[tid] = m; __syncthreads();
  for (int o = 128; o > 0; o >>= 1) { if (tid < o) red[tid] = fmaxf(red[tid], red[tid + o]); __syncthreads(); }
  m = red[0];
  float z = 0.f, ss = 0.f, dn = 0.f, dd = 0.f;
  for (int k = tid; k < TB_; k += 256) {
    float s = srow[k], d = drow[k];
    float e = expf(s - m);
    z += e; ss += e * s; dn += e * d; dd += d;
  }
  red4[tid] = make_float4(z, ss, dn, dd); __syncthreads();
  for (int o = 128; o > 0; o >>= 1) {
    if (tid < o) {
      red4[tid].x += red4[tid + o].x; red4[tid].y += red4[tid + o].y;
      red4[tid].z += red4[tid + o].z; red4[tid].w += red4[tid + o].w;
    }
    __syncthreads();
  }
  if (tid == 0) {
    z = red4[0].x; ss = red4[0].y; dn = red4[0].z; dd = red4[0].w;
    const float lz = logf(z);
    const size_t o_ = (size_t)b * TB_ + q;
    ent_raw[o_] = ((m + lz) - ss / z) * (1.0f / logf(1024.0f));
    dep[o_]     = (dn / z) / dd;
    rowm[o_]    = m;
    rowrz[o_]   = 1.0f / z;
  }
}

__global__ void __launch_bounds__(256)
entnorm_kernel(const float* __restrict__ ent_raw, float* __restrict__ entn)
{
  __shared__ float red[256];
  const int b = blockIdx.x, tid = threadIdx.x;
  float s = 0.f;
  for (int q = tid; q < TB_; q += 256) s += fabsf(ent_raw[(size_t)b * TB_ + q]);
  red[tid] = s; __syncthreads();
  for (int o = 128; o > 0; o >>= 1) { if (tid < o) red[tid] += red[tid + o]; __syncthreads(); }
  const float inv = 1.0f / fmaxf(red[0], 1e-12f);
  for (int q = tid; q < TB_; q += 256)
    entn[(size_t)b * TB_ + q] = ent_raw[(size_t)b * TB_ + q] * inv;
}

// attw'[b,q,k] = softmax(sim)[q,k] * (1 + dep[b,k])   (bf16, feeds ctx GEMM)
__global__ void __launch_bounds__(256)
attw_kernel(const float* __restrict__ sim, const float* __restrict__ rowm,
            const float* __restrict__ rowrz, const float* __restrict__ dep,
            bf16* __restrict__ attw)
{
  size_t idx = (size_t)blockIdx.x * 256 + threadIdx.x;   // 16 * 1024 * 1024
  int b = (int)(idx >> 20);
  int q = (int)((idx >> 10) & 1023);
  int k = (int)(idx & 1023);
  size_t rq = (size_t)b * TB_ + q;
  float p = expf(sim[idx] - rowm[rq]) * rowrz[rq];
  attw[idx] = f2bf(p * (1.0f + dep[(size_t)b * TB_ + k]));
}

// -------- host-side GEMM dispatch helper ------------------------------------
template<int MODE>
static void gemm(hipStream_t st,
                 const bf16* A, int lda, long long sAi, long long sAj,
                 const bf16* B, int ldb, long long sBi, long long sBj,
                 void* C, int ldc, long long sCi, long long sCj,
                 int M, int N, int K, int batch,
                 const float* addend = nullptr, int ld_add = 0,
                 const float* rv0 = nullptr, const float* rv1 = nullptr,
                 long long sRi = 0, long long sRj = 0,
                 const float* cv0 = nullptr, const float* cv1 = nullptr,
                 int cstride = 0)
{
  dim3 g(N / 128, M / 64, batch), blk(128);
  float* Cf = (MODE == 0 || MODE == 3) ? (float*)C : nullptr;
  bf16*  Cb = (MODE == 0 || MODE == 3) ? nullptr   : (bf16*)C;
  gemm_bf16_nt<MODE><<<g, blk, 0, st>>>(A, lda, sAi, sAj, B, ldb, sBi, sBj,
                                        Cf, Cb, ldc, sCi, sCj, M, N, K,
                                        addend, ld_add, rv0, rv1, sRi, sRj,
                                        cv0, cv1, cstride);
}

extern "C" void kernel_launch(void* const* d_in, const int* in_sizes, int n_in,
                              void* d_out, int out_size, void* d_ws, size_t ws_size,
                              hipStream_t stream)
{
  (void)in_sizes; (void)n_in; (void)out_size; (void)ws_size;
  const float* x    = (const float*)d_in[0];   // [4096,1024]
  const float* Wq   = (const float*)d_in[1];   // [1,1024,1024]
  const float* Wk   = (const float*)d_in[2];
  const float* Wv   = (const float*)d_in[3];
  const float* Wout = (const float*)d_in[4];   // [1024,1024]
  const float* W2   = (const float*)d_in[5];   // [256,258]
  float* out = (float*)d_out;                  // [4096,1024] f32

  // ---- workspace layout (256B aligned) ----
  char* p = (char*)d_ws;
  auto alloc = [&](size_t bytes) -> char* {
    char* r = p; p += (bytes + 255) & ~(size_t)255; return r;
  };
  bf16* xbf  = (bf16*)alloc((size_t)T_ * D_ * 2);
  bf16* xu   = (bf16*)alloc((size_t)T_ * D_ * 2);
  bf16* Qb   = (bf16*)alloc((size_t)T_ * D_ * 2);
  bf16* Kb   = (bf16*)alloc((size_t)T_ * D_ * 2);
  bf16* Vt   = (bf16*)alloc((size_t)T_ * D_ * 2);   // transposed: [D, T]
  bf16* Wqb  = (bf16*)alloc((size_t)D_ * D_ * 2);
  bf16* Wkb  = (bf16*)alloc((size_t)D_ * D_ * 2);
  bf16* Wvb  = (bf16*)alloc((size_t)D_ * D_ * 2);
  bf16* Wob  = (bf16*)alloc((size_t)D_ * D_ * 2);
  bf16* W2a  = (bf16*)alloc((size_t)WB_ * WB_ * 2);
  float* AP  = (float*)alloc((size_t)TB_ * TB_ * 4);
  float* Dd  = (float*)alloc((size_t)S_ * TB_ * TB_ * 4);
  float* sim = (float*)alloc((size_t)NB_ * TB_ * TB_ * 4);
  bf16*  atw = (bf16*)alloc((size_t)NB_ * TB_ * TB_ * 2);
  bf16*  ctx = (bf16*)alloc((size_t)NB_ * TB_ * WB_ * 2);
  bf16*  Ybf = (bf16*)alloc((size_t)T_ * D_ * 2);
  float* entr= (float*)alloc((size_t)NB_ * TB_ * 4);
  float* entn= (float*)alloc((size_t)NB_ * TB_ * 4);
  float* dep = (float*)alloc((size_t)NB_ * TB_ * 4);
  float* rowm= (float*)alloc((size_t)NB_ * TB_ * 4);
  float* rowz= (float*)alloc((size_t)NB_ * TB_ * 4);

  const long long MB1 = (long long)TB_ * TB_;        // 1M elements
  // ---- precompute / conversions ----
  convert_kernel<<<4096, 256, 0, stream>>>(Wq,   Wqb, D_ * D_);
  convert_kernel<<<4096, 256, 0, stream>>>(Wk,   Wkb, D_ * D_);
  convert_kernel<<<4096, 256, 0, stream>>>(Wv,   Wvb, D_ * D_);
  convert_kernel<<<4096, 256, 0, stream>>>(Wout, Wob, D_ * D_);
  w2a_kernel<<<256, 256, 0, stream>>>(W2, W2a);
  ap_kernel<<<4096, 256, 0, stream>>>(AP);
  rownorm_kernel<<<T_, 256, 0, stream>>>(x, xbf, xu);

  // ---- Q = x Wq^T, K = x Wk^T (bf16); V stored transposed [D,T] ----
  gemm<1>(stream, xbf, D_, 0, 0, Wqb, D_, 0, 0, Qb, D_, 0, 0, T_, D_, D_, 1);
  gemm<1>(stream, xbf, D_, 0, 0, Wkb, D_, 0, 0, Kb, D_, 0, 0, T_, D_, D_, 1);
  gemm<2>(stream, xbf, D_, 0, 0, Wvb, D_, 0, 0, Vt, T_, 0, 0, T_, D_, D_, 1);

  // ---- diagonal diversity blocks: Dd[i] = 1 - xu_i xu_i^T (batch over j=z) --
  gemm<3>(stream, xu, D_, 0, (long long)TB_ * D_,
                  xu, D_, 0, (long long)TB_ * D_,
                  Dd, TB_, 0, MB1, TB_, TB_, D_, S_);

  // ---- sim[b] = qb kb^T + AP, b=(i,j) over 16 blocks ----
  gemm<0>(stream, Qb, D_, (long long)TB_ * D_, WB_,
                  Kb, D_, (long long)TB_ * D_, WB_,
                  sim, TB_, 4 * MB1, MB1,
                  TB_, TB_, WB_, NB_, AP, TB_);

  // ---- softmax statistics, entropy, dependency ----
  { dim3 g(TB_, NB_); softstat_kernel<<<g, 256, 0, stream>>>(sim, Dd, entr, dep, rowm, rowz); }
  entnorm_kernel<<<NB_, 256, 0, stream>>>(entr, entn);
  attw_kernel<<<65536, 256, 0, stream>>>(sim, rowm, rowz, dep, atw);

  // ---- ctx[b] = attw'[b] @ vb  (B operand = Vt slice, K-contiguous) ----
  gemm<1>(stream, atw, TB_, 4 * MB1, MB1,
                  Vt, T_, (long long)TB_, (long long)WB_ * T_,
                  ctx, WB_, 4LL * TB_ * WB_, (long long)TB_ * WB_,
                  TB_, WB_, TB_, NB_);

  // ---- y[b] = ctx[b] W2a^T + ent[q] W2[:,256] + dep[q] W2[:,257] -> Y ----
  gemm<5>(stream, ctx, WB_, 4LL * TB_ * WB_, (long long)TB_ * WB_,
                  W2a, WB_, 0, 0,
                  Ybf, D_, (long long)TB_ * D_, (long long)WB_,
                  TB_, WB_, WB_, NB_,
                  nullptr, 0,
                  entn, dep, 4LL * TB_, (long long)TB_,
                  W2 + 256, W2 + 257, 258);

  // ---- out = Y Wout^T (f32) ----
  gemm<0>(stream, Ybf, D_, 0, 0, Wob, D_, 0, 0, out, D_, 0, 0, T_, D_, D_, 1);
}
